// SelfAttention_36593121361958
// MI455X (gfx1250) — matrix operand order repaired and checked by reading, not compiled, not run on hardware
//
#include <hip/hip_runtime.h>
#include <hip/hip_bf16.h>

#define B_  8
#define T_  1024
#define E_  128
#define H_  8
#define HE_ (H_ * E_)   // 1024

typedef __attribute__((ext_vector_type(16))) _Float16 v16h;
typedef __attribute__((ext_vector_type(8)))  float    v8f;

// ---- WMMA fragment loaders (per CDNA5 ISA 7.12.2 layouts, wave32) ----
// A: 16x32 (MxK), row-major source with leading dim lda.
// lanes 0-15: M=lane, VGPR j holds K = (j>=4?16:0) + (j&3)*2 (+1)
// lanes16-31: same M, K offset +8 (and +24 for j>=4)
__device__ __forceinline__ v16h load_frag_a(const _Float16* A, int lda) {
    const int lane = threadIdx.x & 31;
    const int l = lane & 15, hi = lane >> 4;
    const _Float16* row = A + (size_t)l * lda;
    v16h a;
#pragma unroll
    for (int j = 0; j < 8; ++j) {
        int k = ((j >> 2) << 4) + (hi << 3) + ((j & 3) << 1);
        a[2 * j]     = row[k];
        a[2 * j + 1] = row[k + 1];
    }
    return a;
}

// B: 32x16 (KxN). Source is BT = N x K row-major (leading dim ldb), i.e.
// B[k][n] = BT[n*ldb + k].  lanes 0-15: N=lane, VGPR j holds K = (j>=4?16:0) + (j&3)*2 (+1);
// lanes 16-31: K offset +8.
__device__ __forceinline__ v16h load_frag_b(const _Float16* BT, int ldb) {
    const int lane = threadIdx.x & 31;
    const int l = lane & 15, hi = lane >> 4;
    const _Float16* row = BT + (size_t)l * ldb;
    v16h b;
#pragma unroll
    for (int j = 0; j < 8; ++j) {
        int k = ((j >> 2) << 4) + (hi << 3) + ((j & 3) << 1);
        b[2 * j]     = row[k];
        b[2 * j + 1] = row[k + 1];
    }
    return b;
}
// C/D 16x16 f32: lane l holds column n=l, VGPR r holds row m = r + 8*hi.

// ---- 1) fp32 -> f16 conversions (+ weight transposes) ----
__global__ __launch_bounds__(256) void convert_kernel(
        const float* __restrict__ x,  const float* __restrict__ Wq,
        const float* __restrict__ Wk, const float* __restrict__ Wv,
        const float* __restrict__ Wo,
        _Float16* __restrict__ Xh,  _Float16* __restrict__ WqT,
        _Float16* __restrict__ WkT, _Float16* __restrict__ WvT,
        _Float16* __restrict__ WoT) {
    int i = blockIdx.x * blockDim.x + threadIdx.x;
    int sel = blockIdx.y;
    if (sel == 0) {
        if (i < B_ * T_ * E_) Xh[i] = (_Float16)x[i];
    } else if (sel <= 3) {
        if (i < E_ * HE_) {
            const float* W = (sel == 1) ? Wq : (sel == 2) ? Wk : Wv;
            _Float16*  WT  = (sel == 1) ? WqT : (sel == 2) ? WkT : WvT;
            int hd = i / E_, e = i % E_;
            WT[i] = (_Float16)W[(size_t)e * HE_ + hd];   // WT[hd][e]
        }
    } else {
        if (i < HE_ * E_) {
            int n = i / HE_, k = i % HE_;
            WoT[i] = (_Float16)Wo[(size_t)k * E_ + n];   // WoT[n][k]
        }
    }
}

// ---- 2) Q/K/V projection GEMMs.  Q scaled by 1/sqrt(E); V stored transposed. ----
__global__ __launch_bounds__(256) void proj_kernel(
        const _Float16* __restrict__ Xh,
        const _Float16* __restrict__ WqT, const _Float16* __restrict__ WkT,
        const _Float16* __restrict__ WvT,
        _Float16* __restrict__ Q, _Float16* __restrict__ K, _Float16* __restrict__ VT) {
    const int wid  = threadIdx.x >> 5;
    const int lane = threadIdx.x & 31;
    const int l = lane & 15, hi = lane >> 4;
    const int TT = T_ / 16;                       // 64 row tiles
    int gw = blockIdx.x * 8 + wid;                // over B * TT * H
    int h  = gw % H_;
    int tt = (gw / H_) % TT;
    int b  = gw / (H_ * TT);
    int t0 = tt * 16;
    int sel = blockIdx.y;
    const _Float16* WT = (sel == 0) ? WqT : (sel == 1) ? WkT : WvT;

    v16h a[4];
#pragma unroll
    for (int kk = 0; kk < 4; ++kk)
        a[kk] = load_frag_a(Xh + ((size_t)b * T_ + t0) * E_ + kk * 32, E_);

    const size_t bh = (size_t)b * H_ + h;
#pragma unroll
    for (int dd = 0; dd < 8; ++dd) {
        v8f acc = {};
#pragma unroll
        for (int kk = 0; kk < 4; ++kk) {
            v16h bf = load_frag_b(WT + (size_t)(h * E_ + dd * 16) * E_ + kk * 32, E_);
            acc = __builtin_amdgcn_wmma_f32_16x16x32_f16(false, a[kk], false, bf,
                                                         (short)0, acc, false, false);
        }
        if (sel == 0) {                            // Q, fold in 1/sqrt(128)
#pragma unroll
            for (int r = 0; r < 8; ++r) {
                int t = t0 + r + 8 * hi;
                Q[(bh * T_ + t) * E_ + dd * 16 + l] =
                    (_Float16)(acc[r] * 0.08838834764831845f);
            }
        } else if (sel == 1) {                     // K row-major [t][d]
#pragma unroll
            for (int r = 0; r < 8; ++r) {
                int t = t0 + r + 8 * hi;
                K[(bh * T_ + t) * E_ + dd * 16 + l] = (_Float16)acc[r];
            }
        } else {                                   // V transposed [d][t]
#pragma unroll
            for (int r = 0; r < 8; ++r) {
                int t = t0 + r + 8 * hi;
                VT[(bh * E_ + dd * 16 + l) * (size_t)T_ + t] = (_Float16)acc[r];
            }
        }
    }
}

// ---- 3) Flash-style masked attention.  1 workgroup = (b,h) x 128 query rows,
//         1 wave = 16 query rows, streaming 32 keys per iteration. ----
__global__ __launch_bounds__(256) void attn_kernel(
        const _Float16* __restrict__ Q, const _Float16* __restrict__ K,
        const _Float16* __restrict__ VT, const int* __restrict__ mask,
        _Float16* __restrict__ Y) {
    __shared__ _Float16 Plds[8 * 512];             // 16x32 f16 P-tile per wave
    const int wid  = threadIdx.x >> 5;
    const int lane = threadIdx.x & 31;
    const int l = lane & 15, hi = lane >> 4;
    const int bh = blockIdx.y;
    const int b = bh / H_, h = bh % H_;
    const int t0 = blockIdx.x * 128 + wid * 16;

    const _Float16* Qbh  = Q  + (size_t)bh * T_ * E_;
    const _Float16* Kbh  = K  + (size_t)bh * T_ * E_;
    const _Float16* VTbh = VT + (size_t)bh * E_ * T_;
    const int*      Mbh  = mask + (size_t)bh * T_ * T_;

    v16h aq[4];
#pragma unroll
    for (int kk = 0; kk < 4; ++kk)
        aq[kk] = load_frag_a(Qbh + (size_t)t0 * E_ + kk * 32, E_);

    v8f O[8];
    float mrun[8], lrun[8];
    v8f zero = {};
#pragma unroll
    for (int dd = 0; dd < 8; ++dd) O[dd] = zero;
#pragma unroll
    for (int r = 0; r < 8; ++r) { mrun[r] = -1e30f; lrun[r] = 0.0f; }

    _Float16* pw = &Plds[wid * 512];

    for (int s0 = 0; s0 < T_; s0 += 32) {
        // scores: S[16][32] over this key block, f32 accumulate
        v8f c0 = {}, c1 = {};
#pragma unroll
        for (int kk = 0; kk < 4; ++kk) {
            v16h bk0 = load_frag_b(Kbh + (size_t)s0 * E_ + kk * 32, E_);
            v16h bk1 = load_frag_b(Kbh + (size_t)(s0 + 16) * E_ + kk * 32, E_);
            c0 = __builtin_amdgcn_wmma_f32_16x16x32_f16(false, aq[kk], false, bk0,
                                                        (short)0, c0, false, false);
            c1 = __builtin_amdgcn_wmma_f32_16x16x32_f16(false, aq[kk], false, bk1,
                                                        (short)0, c1, false, false);
        }
        // mask: one-shot 268MB stream -> non-temporal loads so it doesn't
        // rinse K/V/Y (which we want resident in the 192MB L2)
#pragma unroll
        for (int r = 0; r < 8; ++r) {
            const int* mp = Mbh + (size_t)(t0 + r + 8 * hi) * T_ + s0;
            int m0v = __builtin_nontemporal_load(mp + l);
            int m1v = __builtin_nontemporal_load(mp + 16 + l);
            c0[r] = (m0v != 0) ? c0[r] : -1e9f;
            c1[r] = (m1v != 0) ? c1[r] : -1e9f;
        }
        // online softmax (rows live across the 16-lane half: xor-reduce 1,2,4,8)
#pragma unroll
        for (int r = 0; r < 8; ++r) {
            float tmax = fmaxf(c0[r], c1[r]);
            tmax = fmaxf(tmax, __shfl_xor(tmax, 1, 32));
            tmax = fmaxf(tmax, __shfl_xor(tmax, 2, 32));
            tmax = fmaxf(tmax, __shfl_xor(tmax, 4, 32));
            tmax = fmaxf(tmax, __shfl_xor(tmax, 8, 32));
            float mnew = fmaxf(mrun[r], tmax);
            float cf   = __expf(mrun[r] - mnew);
            mrun[r] = mnew;
            c0[r] = __expf(c0[r] - mnew);
            c1[r] = __expf(c1[r] - mnew);
            float ts = c0[r] + c1[r];
            ts += __shfl_xor(ts, 1, 32);
            ts += __shfl_xor(ts, 2, 32);
            ts += __shfl_xor(ts, 4, 32);
            ts += __shfl_xor(ts, 8, 32);
            lrun[r] = lrun[r] * cf + ts;
#pragma unroll
            for (int dd = 0; dd < 8; ++dd) O[dd][r] *= cf;
        }
        // P: C-layout -> row-major LDS -> A-layout
#pragma unroll
        for (int r = 0; r < 8; ++r) {
            int row = r + 8 * hi;
            pw[row * 32 + l]      = (_Float16)c0[r];
            pw[row * 32 + 16 + l] = (_Float16)c1[r];
        }
        __syncthreads();
        v16h pa = load_frag_a(pw, 32);
        // O += P @ V   (V stored transposed: contiguous B-fragment loads)
#pragma unroll
        for (int dd = 0; dd < 8; ++dd) {
            v16h bv = load_frag_b(VTbh + (size_t)(dd * 16) * T_ + s0, T_);
            O[dd] = __builtin_amdgcn_wmma_f32_16x16x32_f16(false, pa, false, bv,
                                                           (short)0, O[dd], false, false);
        }
        __syncthreads();
    }
    // normalize, write Y[b][t][h*E + d] in f16 for the output projection
#pragma unroll
    for (int r = 0; r < 8; ++r) {
        float inv = 1.0f / lrun[r];
        int t = t0 + r + 8 * hi;
        _Float16* yr = Y + ((size_t)b * T_ + t) * HE_ + h * E_;
#pragma unroll
        for (int dd = 0; dd < 8; ++dd)
            yr[dd * 16 + l] = (_Float16)(O[dd][r] * inv);
    }
}

// ---- 4) output projection: out = Y(8192x1024) @ Wo + bo, fp32 out ----
__global__ __launch_bounds__(256) void outproj_kernel(
        const _Float16* __restrict__ Y, const _Float16* __restrict__ WoT,
        const float* __restrict__ bo, float* __restrict__ out) {
    const int wid  = threadIdx.x >> 5;
    const int lane = threadIdx.x & 31;
    const int l = lane & 15, hi = lane >> 4;
    int gw = blockIdx.x * 8 + wid;
    int nt = gw & 7;            // E/16 = 8 col tiles
    int mt = gw >> 3;
    int m0 = mt * 16, n0 = nt * 16;
    v8f acc = {};
    for (int k0 = 0; k0 < HE_; k0 += 32) {
        v16h a  = load_frag_a(Y   + (size_t)m0 * HE_ + k0, HE_);
        v16h bf = load_frag_b(WoT + (size_t)n0 * HE_ + k0, HE_);
        acc = __builtin_amdgcn_wmma_f32_16x16x32_f16(false, a, false, bf,
                                                     (short)0, acc, false, false);
    }
    float bias = bo[n0 + l];
#pragma unroll
    for (int r = 0; r < 8; ++r) {
        int m = m0 + r + 8 * hi;
        out[(size_t)m * E_ + n0 + l] = acc[r] + bias;
    }
}

extern "C" void kernel_launch(void* const* d_in, const int* in_sizes, int n_in,
                              void* d_out, int out_size, void* d_ws, size_t ws_size,
                              hipStream_t stream) {
    (void)in_sizes; (void)n_in; (void)out_size; (void)ws_size;
    const float* x    = (const float*)d_in[0];
    const int*   mask = (const int*)  d_in[1];
    const float* Wq   = (const float*)d_in[2];
    const float* Wk   = (const float*)d_in[3];
    const float* Wv   = (const float*)d_in[4];
    const float* Wo   = (const float*)d_in[5];
    const float* bo   = (const float*)d_in[6];

    _Float16* p   = (_Float16*)d_ws;
    _Float16* Xh  = p;  p += (size_t)B_ * T_ * E_;        // 1M
    _Float16* WqT = p;  p += (size_t)E_ * HE_;            // 128K
    _Float16* WkT = p;  p += (size_t)E_ * HE_;
    _Float16* WvT = p;  p += (size_t)E_ * HE_;
    _Float16* WoT = p;  p += (size_t)HE_ * E_;
    _Float16* Qh  = p;  p += (size_t)B_ * H_ * T_ * E_;   // 8M each
    _Float16* Kh  = p;  p += (size_t)B_ * H_ * T_ * E_;
    _Float16* VTh = p;  p += (size_t)B_ * H_ * T_ * E_;
    _Float16* Yh  = p;  p += (size_t)B_ * T_ * HE_;       // ~70 MB total

    convert_kernel<<<dim3((B_ * T_ * E_) / 256, 5), 256, 0, stream>>>(
        x, Wq, Wk, Wv, Wo, Xh, WqT, WkT, WvT, WoT);
    proj_kernel<<<dim3((B_ * (T_ / 16) * H_) / 8, 3), 256, 0, stream>>>(
        Xh, WqT, WkT, WvT, Qh, Kh, VTh);
    attn_kernel<<<dim3(T_ / 128, B_ * H_), 256, 0, stream>>>(
        Qh, Kh, VTh, mask, Yh);
    outproj_kernel<<<dim3((B_ * T_ / 16) * (E_ / 16) / 8), 256, 0, stream>>>(
        Yh, WoT, bo, (float*)d_out);
}
